// MultiScaleVQVAE_ST_45810121179715
// MI455X (gfx1250) — compile-verified, matrix operand order checked by reading, not run on hardware
//
#include <hip/hip_runtime.h>
#include <hip/hip_bf16.h>
#include <math.h>

// ---------------------------------------------------------------------------
// MultiScale VQ-VAE forward for gfx1250 (MI455X), WMMA f16 path, v4b.
//
// B operand tiles (weights / codebook), identical across the 4 waves of a
// block, staged cooperatively into LDS (double-buffered) via
// GLOBAL_LOAD_ASYNC_TO_LDS_B128 + s_wait_asynccnt (builtin-guarded; sync
// reg->LDS fallback otherwise), read back as ds b128 fragments.
//
// Fragment fact (ISA 7.12.2): per lane, fragment VGPRs 0-3 hold
// K = half*8..half*8+7 and VGPRs 4-7 hold K = 16+half*8.. => with f16
// K-contiguous operands every fragment is two aligned b128 loads.
// ---------------------------------------------------------------------------

typedef __attribute__((ext_vector_type(16))) _Float16     v16h;
typedef __attribute__((ext_vector_type(8)))  float        v8f;
typedef __attribute__((ext_vector_type(8)))  unsigned int v8u;
typedef __attribute__((ext_vector_type(4)))  int          v4i;

#define S15 15
#define EDIM 256
#define HDIM 128
#define VSZ 8192
#define GG 32
#define BB 512

#define AS1 __attribute__((address_space(1)))
#define AS3 __attribute__((address_space(3)))

#if defined(__has_builtin)
#if __has_builtin(__builtin_amdgcn_global_load_async_to_lds_b128) && \
    __has_builtin(__builtin_amdgcn_s_wait_asynccnt)
#define USE_ASYNC_LDS 1
#endif
#endif
#ifndef USE_ASYNC_LDS
#define USE_ASYNC_LDS 0
#endif

__device__ __forceinline__ int wmma_k_of(int e, int half) {
  int v = e >> 1, p = e & 1;
  return ((v & 4) ? 16 : 0) + half * 8 + (v & 3) * 2 + p;
}

__device__ __forceinline__ v8u frag_b128(const _Float16* p) {
  const uint4 lo = *(const uint4*)(p);
  const uint4 hi = *(const uint4*)(p + 16);
  v8u u = {lo.x, lo.y, lo.z, lo.w, hi.x, hi.y, hi.z, hi.w};
  return u;
}

// one 16-byte global -> LDS transfer (async when available)
__device__ __forceinline__ void stage_b128(const _Float16* src, _Float16* dst_lds) {
#if USE_ASYNC_LDS
  __builtin_amdgcn_global_load_async_to_lds_b128((AS1 v4i*)src, (AS3 v4i*)dst_lds, 0, 0);
#else
  *(uint4*)dst_lds = *(const uint4*)src;
#endif
}
__device__ __forceinline__ void stage_fence() {
#if USE_ASYNC_LDS
  __builtin_amdgcn_s_wait_asynccnt(0);
#endif
  __syncthreads();
}

#define LDSB_STRIDE 40   // 64x32 weight tile rows padded 32->40 halves (bank skew)
#define LDSC_STRIDE 264  // 16x256 codebook tile rows padded 256->264 halves

// ---------------------------------------------------------------------------
// Implicit-GEMM conv (ks = 1 or 3). One wave -> 16x64 tile, 4 waves/block.
// B tile (64 x 32 f16, shared by all 4 waves) staged in LDS, double-buffered.
// ---------------------------------------------------------------------------
__global__ void gemm_conv_wmma(const _Float16* __restrict__ act16, // NHWC f16
                               const _Float16* __restrict__ cb16,  // gather rows
                               const _Float16* __restrict__ wh,    // [Cp,Kp] f16 k'-order
                               const float*    __restrict__ bias,
                               const int*      __restrict__ gidx,
                               const float*    __restrict__ resid, // f32 NCHW
                               float*    __restrict__ out,          // f32 NCHW
                               _Float16* __restrict__ out16,        // f16 [M,Cout] or null
                               int Bn, int Cin, int Cout, int H, int W, int ks, int Kp)
{
  __shared__ __attribute__((aligned(16))) _Float16 lsB[2][64 * LDSB_STRIDE];

  const int tid  = threadIdx.x;
  const int lane = tid & 31;
  const int wv   = tid >> 5;
  const int HW  = H * W;
  const int M   = Bn * HW;
  const int K   = Cin * ks * ks;
  const int pad = ks >> 1;
  const int mt  = (M + 15) >> 4;
  int mtile = blockIdx.x * 4 + wv;
  if (mtile >= mt) mtile = mt - 1;             // keep all waves for barriers

  const int half = lane >> 4;
  const int l16  = lane & 15;

  const int  am     = mtile * 16 + l16;
  const bool avalid = (am < M);
  const int  ab = avalid ? am / HW : 0;
  const int  as = avalid ? am % HW : 0;
  const int  ay = as / W;
  const int  ax = as % W;
  long arow = 0;
  if (gidx && avalid) arow = (long)gidx[am] * Kp;

  const bool vecA  = (Cin % 8) == 0;
  const int  nbase = blockIdx.y * 64;

  // cooperative stage: thread -> (row = tid>>1, seg = tid&1), 1 b128 each
  const int srow = tid >> 1, sseg = tid & 1;
  const _Float16* sgsrc = wh + (long)(nbase + srow) * Kp + sseg * 16;
  const int sdst = srow * LDSB_STRIDE + sseg * 16;

  stage_b128(sgsrc, &lsB[0][sdst]);
  stage_fence();

  v8f acc0 = {}, acc1 = {}, acc2 = {}, acc3 = {};
  int cur = 0;
  for (int kt = 0; kt < Kp; kt += 32) {
    if (kt + 32 < Kp) stage_b128(sgsrc + kt + 32, &lsB[cur ^ 1][sdst]);

    // ---- A fragment: two 16-byte K-runs per lane (global b128) ----
    v16h af;
    if (gidx) {
      v8u au = {};
      if (avalid) au = frag_b128(cb16 + arow + kt + half * 8);
      af = __builtin_bit_cast(v16h, au);
    } else if (vecA) {
      v8u au = {};
#pragma unroll
      for (int seg = 0; seg < 2; ++seg) {
        const int kk = kt + half * 8 + seg * 16;
        if (avalid && kk < K) {
          const int pix = kk / Cin, cio = kk % Cin;     // run stays in one pixel
          const int ky = pix / ks, kx = pix - ky * ks;
          const int iy = ay + ky - pad, ix = ax + kx - pad;
          if (iy >= 0 && iy < H && ix >= 0 && ix < W) {
            const uint4 v = *(const uint4*)(act16 + (((long)ab * H + iy) * W + ix) * Cin + cio);
            au[seg * 4 + 0] = v.x; au[seg * 4 + 1] = v.y;
            au[seg * 4 + 2] = v.z; au[seg * 4 + 3] = v.w;
          }
        }
      }
      af = __builtin_bit_cast(v16h, au);
    } else {                                            // Cin==1 fallback
#pragma unroll
      for (int e = 0; e < 16; ++e) {
        const int k = kt + wmma_k_of(e, half);
        _Float16 av = (_Float16)0.f;
        if (avalid && k < K) {
          const int pix = k / Cin, cio = k % Cin;
          const int ky = pix / ks, kx = pix - ky * ks;
          const int iy = ay + ky - pad, ix = ax + kx - pad;
          if (iy >= 0 && iy < H && ix >= 0 && ix < W)
            av = act16[(((long)ab * H + iy) * W + ix) * Cin + cio];
        }
        af[e] = av;
      }
    }

    // ---- 4 B fragments from LDS (ds b128), 4 WMMAs, A reused ----
#pragma unroll
    for (int j = 0; j < 4; ++j) {
      const v8u bu = frag_b128(&lsB[cur][(j * 16 + l16) * LDSB_STRIDE + half * 8]);
      const v16h bf = __builtin_bit_cast(v16h, bu);
      v8f& acc = (j == 0) ? acc0 : (j == 1) ? acc1 : (j == 2) ? acc2 : acc3;
      acc = __builtin_amdgcn_wmma_f32_16x16x32_f16(false, af, false, bf,
                                                   (short)0, acc, false, false);
    }

    stage_fence();          // async copies done + LDS reads of cur complete
    cur ^= 1;
  }

  // D layout: VGPR r holds row (r + half*8), col = l16
#pragma unroll
  for (int j = 0; j < 4; ++j) {
    const int n = nbase + j * 16 + l16;
    if (n < Cout) {
      const float bsv = bias ? bias[n] : 0.f;
      const v8f acc = (j == 0) ? acc0 : (j == 1) ? acc1 : (j == 2) ? acc2 : acc3;
#pragma unroll
      for (int r = 0; r < 8; ++r) {
        const int m = mtile * 16 + r + half * 8;
        if (m < M) {
          const int b = m / HW, s = m % HW;
          const long oi = ((long)b * Cout + n) * HW + s;
          float v = acc[r] + bsv;
          if (resid) v += resid[oi];
          out[oi] = v;
          if (out16) out16[(long)m * Cout + n] = (_Float16)v;
        }
      }
    }
  }
}

// ---------------------------------------------------------------------------
// VQ nearest-code search: argmin_n (||c_n||^2 - 2 x_m . c_n).
// A rows hoisted to registers; codebook tiles (16x256 f16, shared by the 4
// waves of the block) staged in LDS, double-buffered, async when available.
// ---------------------------------------------------------------------------
__global__ void vq_argmin_wmma(const _Float16* __restrict__ q16,   // [M,256] f16
                               const _Float16* __restrict__ cb16,  // [V,256] f16
                               const float*    __restrict__ cnorm, // [V]
                               int* __restrict__ idx_out, int M)
{
  __shared__ __attribute__((aligned(16))) _Float16 lsC[2][16 * LDSC_STRIDE];

  const int tid  = threadIdx.x;
  const int lane = tid & 31;
  const int wv   = tid >> 5;
  const int mt   = (M + 15) >> 4;
  int mtile = blockIdx.x * 4 + wv;
  if (mtile >= mt) mtile = mt - 1;             // keep all waves for barriers
  const int half = lane >> 4;
  const int l16  = lane & 15;

  const int  am = mtile * 16 + l16;
  const bool avalid = (am < M);
  const _Float16* qrow = q16 + (long)am * EDIM;

  v16h afr[8];
#pragma unroll
  for (int kt = 0; kt < 8; ++kt) {
    v8u au = {};
    if (avalid) au = frag_b128(qrow + kt * 32 + half * 8);
    afr[kt] = __builtin_bit_cast(v16h, au);
  }

  // cooperative stage of one 16x256 tile: 512 b128 chunks, 4 per thread
  auto stageC = [&](int nt, int buf) {
#pragma unroll
    for (int c = 0; c < 4; ++c) {
      const int chunk = c * 128 + tid;
      const int row = chunk >> 4, seg = chunk & 15;
      stage_b128(cb16 + ((long)nt * 16 + row) * EDIM + seg * 16,
                 &lsC[buf][row * LDSC_STRIDE + seg * 16]);
    }
  };

  float best[8];
  int   bidx[8];
#pragma unroll
  for (int r = 0; r < 8; ++r) { best[r] = 3.4e38f; bidx[r] = 0; }

  stageC(0, 0);
  stage_fence();
  int cur = 0;
  for (int nt = 0; nt < VSZ / 16; ++nt) {
    if (nt + 1 < VSZ / 16) stageC(nt + 1, cur ^ 1);

    const _Float16* crow = &lsC[cur][l16 * LDSC_STRIDE];
    v8f acc = {};
#pragma unroll
    for (int kt = 0; kt < 8; ++kt) {
      const v8u bu = frag_b128(crow + kt * 32 + half * 8);
      const v16h bf = __builtin_bit_cast(v16h, bu);
      acc = __builtin_amdgcn_wmma_f32_16x16x32_f16(false, afr[kt], false, bf,
                                                   (short)0, acc, false, false);
    }
    const int   n  = nt * 16 + l16;
    const float cn = cnorm[n];
#pragma unroll
    for (int r = 0; r < 8; ++r) {
      const float sc = cn - 2.0f * acc[r];
      if (sc < best[r]) { best[r] = sc; bidx[r] = n; }
    }

    stage_fence();
    cur ^= 1;
  }

#pragma unroll
  for (int off = 1; off < 16; off <<= 1) {
#pragma unroll
    for (int r = 0; r < 8; ++r) {
      const float ob = __shfl_xor(best[r], off, 16);
      const int   oi = __shfl_xor(bidx[r], off, 16);
      if (ob < best[r] || (ob == best[r] && oi < bidx[r])) { best[r] = ob; bidx[r] = oi; }
    }
  }
  if (l16 == 0 && avalid) {
#pragma unroll
    for (int r = 0; r < 8; ++r) {
      const int m = mtile * 16 + r + half * 8;
      if (m < M) idx_out[m] = bidx[r];
    }
  }
}

// f32 weights [Cout][Cin][ks][ks] -> f16 [Cp,Kp], k' = (ky*ks+kx)*Cin + ci.
__global__ void wcvt_kernel(const float* __restrict__ w, _Float16* __restrict__ wh,
                            int Cout, int Cin, int ks, int Kp, long tot) {
  const long t = (long)blockIdx.x * blockDim.x + threadIdx.x;
  if (t >= tot) return;
  const int n = (int)(t / Kp), kp = (int)(t % Kp);
  const int K = Cin * ks * ks;
  _Float16 v = (_Float16)0.f;
  if (n < Cout && kp < K) {
    const int pix = kp / Cin, ci = kp % Cin;
    v = (_Float16)w[(long)n * K + ci * ks * ks + pix];
  }
  wh[t] = v;
}

__global__ void cnorm_kernel(const float* __restrict__ cb, float* __restrict__ cn) {
  const int v = blockIdx.x * blockDim.x + threadIdx.x;
  if (v >= VSZ) return;
  const float* row = cb + (long)v * EDIM;
  float s = 0.f;
  for (int k = 0; k < EDIM; ++k) { const float x = row[k]; s += x * x; }
  cn[v] = s;
}

// GroupNorm(32) + SiLU: f32 NCHW in; f16 NHWC out (+ optional f32 NCHW copy).
__global__ void gn_silu_kernel(const float* __restrict__ x,
                               const float* __restrict__ gam,
                               const float* __restrict__ bet,
                               _Float16* __restrict__ y16,  // NHWC
                               float*    __restrict__ yf,   // NCHW or null
                               int C, int HW)
{
  const int b = blockIdx.x / GG;
  const int g = blockIdx.x % GG;
  const int cpg = C / GG;
  const int n = cpg * HW;
  const float* base = x + ((long)b * C + g * cpg) * HW;
  __shared__ float ssum[256], ssq[256];
  float s = 0.f, q = 0.f;
  for (int i = threadIdx.x; i < n; i += blockDim.x) {
    const float v = base[i]; s += v; q += v * v;
  }
  ssum[threadIdx.x] = s; ssq[threadIdx.x] = q;
  __syncthreads();
  for (int off = blockDim.x >> 1; off > 0; off >>= 1) {
    if ((int)threadIdx.x < off) {
      ssum[threadIdx.x] += ssum[threadIdx.x + off];
      ssq[threadIdx.x]  += ssq[threadIdx.x + off];
    }
    __syncthreads();
  }
  const float mu   = ssum[0] / n;
  const float var  = ssq[0] / n - mu * mu;
  const float rstd = rsqrtf(var + 1e-5f);
  float* yfb = yf ? yf + ((long)b * C + g * cpg) * HW : nullptr;
  for (int i = threadIdx.x; i < n; i += blockDim.x) {
    const int cl = i / HW, sp = i - cl * HW;
    const int c = g * cpg + cl;
    const float t = (base[i] - mu) * rstd * gam[c] + bet[c];
    const float v = t / (1.f + expf(-t));               // SiLU
    if (yfb) yfb[i] = v;
    y16[((long)b * HW + sp) * C + c] = (_Float16)v;
  }
}

// torch adaptive_avg_pool2d window rule -> f16 (Cin=1 so NHWC==NCHW)
__global__ void adapool_kernel(const float* __restrict__ x, _Float16* __restrict__ y, int pn) {
  const int t = blockIdx.x * blockDim.x + threadIdx.x;
  const int tot = BB * pn * pn;
  if (t >= tot) return;
  const int b = t / (pn * pn);
  const int r = t % (pn * pn);
  const int i = r / pn, j = r % pn;
  const int si = (i * S15) / pn, ei = ((i + 1) * S15 + pn - 1) / pn;
  const int sj = (j * S15) / pn, ej = ((j + 1) * S15 + pn - 1) / pn;
  const float* xb = x + (long)b * S15 * S15;
  float s = 0.f;
  for (int yy = si; yy < ei; ++yy)
    for (int xx = sj; xx < ej; ++xx)
      s += xb[yy * S15 + xx];
  y[t] = (_Float16)(s / (float)((ei - si) * (ej - sj)));
}

// bilinear upsample (half-pixel, clamped) f32 NCHW -> f16 NHWC channel slice
__global__ void resize_bilinear_kernel(const float* __restrict__ in, _Float16* __restrict__ fused16,
                                       int C, int pn, int chOff, int Cf)
{
  const long t = (long)blockIdx.x * blockDim.x + threadIdx.x;
  const long tot = (long)BB * C * S15 * S15;
  if (t >= tot) return;
  const int x = (int)(t % S15); long r = t / S15;
  const int y = (int)(r % S15); r /= S15;
  const int c = (int)(r % C);
  const int b = (int)(r / C);
  const float scale = (float)pn / (float)S15;
  const float fy = (y + 0.5f) * scale - 0.5f;
  const float fx = (x + 0.5f) * scale - 0.5f;
  const int y0 = (int)floorf(fy), x0 = (int)floorf(fx);
  const float wy = fy - y0, wx = fx - x0;
  const int y0c = min(max(y0, 0), pn - 1), y1c = min(max(y0 + 1, 0), pn - 1);
  const int x0c = min(max(x0, 0), pn - 1), x1c = min(max(x0 + 1, 0), pn - 1);
  const float* p = in + ((long)b * C + c) * pn * pn;
  const float v00 = p[y0c * pn + x0c], v01 = p[y0c * pn + x1c];
  const float v10 = p[y1c * pn + x0c], v11 = p[y1c * pn + x1c];
  const float v = v00 * (1 - wy) * (1 - wx) + v01 * (1 - wy) * wx
                + v10 * wy * (1 - wx)       + v11 * wy * wx;
  fused16[((long)b * (S15 * S15) + y * S15 + x) * Cf + chOff + c] = (_Float16)v;
}

// ---------------------------------------------------------------------------
// Host orchestration. d_in layout (jax pytree: dict keys sorted, lists/tuples
// in order): 0 img; 1..5 codebooks; decoder 6..37; encoders 38+24e (e=0..4);
// post_quant 158+2i; quant 168+2i  (see round-1 map).
// ---------------------------------------------------------------------------
extern "C" void kernel_launch(void* const* d_in, const int* in_sizes, int n_in,
                              void* d_out, int out_size, void* d_ws, size_t ws_size,
                              hipStream_t stream) {
  if (n_in < 178) return;

  const size_t F_BUF    = (size_t)BB * EDIM * S15 * S15;      // 29,491,200
  const size_t H_FUSED  = (size_t)BB * S15 * S15 * 5 * EDIM;  // 147,456,000 halves
  const size_t H_ARENA  = 20u * 1024 * 1024;
  const size_t needB = 4 * F_BUF * sizeof(float)
                     + (H_FUSED + 2 * F_BUF + (size_t)BB * 25 * EDIM + BB * 25 + H_ARENA) * sizeof(_Float16)
                     + VSZ * sizeof(float) + (size_t)BB * 25 * sizeof(int);
  if (ws_size < needB) return;

  float* t0 = (float*)d_ws;
  float* t1 = t0 + F_BUF;
  float* t2 = t1 + F_BUF;
  float* t3 = t2 + F_BUF;
  _Float16* fused16  = (_Float16*)(t3 + F_BUF);
  _Float16* g16a     = fused16 + H_FUSED;
  _Float16* g16b     = g16a + F_BUF;
  _Float16* q16      = g16b + F_BUF;
  _Float16* pooled16 = q16 + (size_t)BB * 25 * EDIM;
  _Float16* arena    = pooled16 + BB * 25;
  float*    cnormb   = (float*)(arena + H_ARENA);
  int*      idxb     = (int*)(cnormb + VSZ);
  size_t aoff = 0;
  auto halloc = [&](size_t n) {
    _Float16* p = arena + aoff;
    aoff += (n + 63) & ~(size_t)63;
    return p;
  };

  const float* X = (const float*)d_in[0];

  auto prep = [&](int widx, int Cout, int Cin, int ks, int& Kp, int& Cp) {
    const int K = Cin * ks * ks;
    Kp = (K + 31) & ~31;
    Cp = ((Cout + 63) / 64) * 64;
    _Float16* wh = halloc((size_t)Cp * Kp);
    const long tot = (long)Cp * Kp;
    wcvt_kernel<<<(int)((tot + 255) / 256), 256, 0, stream>>>(
        (const float*)d_in[widx], wh, Cout, Cin, ks, Kp, tot);
    return wh;
  };

  auto conv = [&](const _Float16* in16, int widx, const float* res, float* out,
                  _Float16* out16, int Cin, int Cout, int H, int W, int ks) {
    int Kp, Cp;
    const _Float16* wh = prep(widx, Cout, Cin, ks, Kp, Cp);
    const int M = BB * H * W, mt = (M + 15) / 16;
    dim3 grid((mt + 3) / 4, Cp / 64);
    gemm_conv_wmma<<<grid, 128, 0, stream>>>(
        in16, nullptr, wh, (const float*)d_in[widx + 1], nullptr, res, out, out16,
        BB, Cin, Cout, H, W, ks, Kp);
  };
  auto gns = [&](const float* x, int gi, _Float16* y16, float* yf, int C, int HW) {
    gn_silu_kernel<<<BB * GG, 256, 0, stream>>>(
        x, (const float*)d_in[gi], (const float*)d_in[gi + 1], y16, yf, C, HW);
  };

  // ---------------- encoders + VQ + post-quant + resize ----------------
  for (int i = 0; i < 5; ++i) {
    const int pn = i + 1, hw = pn * pn;
    const int E  = 38 + 24 * i;
    const float* cb = (const float*)d_in[1 + i];

    _Float16* cb16 = halloc((size_t)VSZ * EDIM);
    {
      const long tot = (long)VSZ * EDIM;
      wcvt_kernel<<<(int)((tot + 255) / 256), 256, 0, stream>>>(cb, cb16, VSZ, EDIM, 1, EDIM, tot);
    }

    adapool_kernel<<<(BB * hw + 255) / 256, 256, 0, stream>>>(X, pooled16, pn);

    conv(pooled16, E + 16, nullptr, t0, nullptr, 1, HDIM, pn, pn, 3);    // conv_in
    gns(t0, E + 20, g16a, t3, HDIM, hw);              // x: f16=g16a, f32=t3
    // resblock 0
    gns(t3, E + 4, g16b, nullptr, HDIM, hw);
    conv(g16b, E + 0, nullptr, t1, nullptr, HDIM, HDIM, pn, pn, 3);
    gns(t1, E + 6, g16b, nullptr, HDIM, hw);
    conv(g16b, E + 2, /*res=*/t3, t0, nullptr, HDIM, HDIM, pn, pn, 3);   // cur=t0
    // resblock 1
    gns(t0, E + 12, g16a, nullptr, HDIM, hw);
    conv(g16a, E + 8, nullptr, t1, nullptr, HDIM, HDIM, pn, pn, 3);
    gns(t1, E + 14, g16b, nullptr, HDIM, hw);
    conv(g16b, E + 10, /*res=*/t0, t1, g16a, HDIM, HDIM, pn, pn, 3);     // cur f16=g16a
    // conv_out + gn_out + silu + quant_conv
    conv(g16a, E + 18, nullptr, t0, nullptr, HDIM, EDIM, pn, pn, 3);
    gns(t0, E + 22, g16b, nullptr, EDIM, hw);
    conv(g16b, 168 + 2 * i, nullptr, t1, q16, EDIM, EDIM, pn, pn, 1);    // q16

    cnorm_kernel<<<(VSZ + 255) / 256, 256, 0, stream>>>(cb, cnormb);
    {
      const int M = BB * hw, mt = (M + 15) / 16;
      vq_argmin_wmma<<<(mt + 3) / 4, 128, 0, stream>>>(q16, cb16, cnormb, idxb, M);
    }
    // post-quant 1x1 conv fused with codebook gather
    {
      int Kp, Cp;
      const _Float16* wh = prep(158 + 2 * i, EDIM, EDIM, 1, Kp, Cp);
      const int M = BB * hw, mt = (M + 15) / 16;
      dim3 grid((mt + 3) / 4, Cp / 64);
      gemm_conv_wmma<<<grid, 128, 0, stream>>>(
          nullptr, cb16, wh, (const float*)d_in[159 + 2 * i], idxb, nullptr, t1, nullptr,
          BB, EDIM, EDIM, pn, pn, 1, Kp);
    }
    {
      const long tot = (long)BB * EDIM * S15 * S15;
      resize_bilinear_kernel<<<(int)((tot + 255) / 256), 256, 0, stream>>>(
          t1, fused16, EDIM, pn, EDIM * i, 5 * EDIM);
    }
  }

  // ---------------- decoder ----------------
  conv(fused16, 32, nullptr, t0, nullptr, 5 * EDIM, 2 * HDIM, S15, S15, 1); // conv_in
  gns(t0, 36, g16a, t3, 2 * HDIM, 225);               // x: f16=g16a, f32=t3
  // block0 (256->128, 1x1 shortcut)
  gns(t3, 10, g16b, nullptr, 2 * HDIM, 225);
  conv(g16b, 6, nullptr, t0, nullptr, 2 * HDIM, HDIM, S15, S15, 3);
  gns(t0, 12, g16b, nullptr, HDIM, 225);
  conv(g16a, 14, nullptr, t2, nullptr, 2 * HDIM, HDIM, S15, S15, 1);        // shortcut
  conv(g16b, 8, /*res=*/t2, t0, nullptr, HDIM, HDIM, S15, S15, 3);          // cur=t0
  // block1
  gns(t0, 20, g16a, nullptr, HDIM, 225);
  conv(g16a, 16, nullptr, t1, nullptr, HDIM, HDIM, S15, S15, 3);
  gns(t1, 22, g16b, nullptr, HDIM, 225);
  conv(g16b, 18, /*res=*/t0, t1, nullptr, HDIM, HDIM, S15, S15, 3);         // cur=t1
  // block2
  gns(t1, 28, g16a, nullptr, HDIM, 225);
  conv(g16a, 24, nullptr, t0, nullptr, HDIM, HDIM, S15, S15, 3);
  gns(t0, 30, g16b, nullptr, HDIM, 225);
  conv(g16b, 26, /*res=*/t1, t0, g16a, HDIM, HDIM, S15, S15, 3);            // cur f16=g16a
  // conv_out 3x3 128->1 straight into d_out
  conv(g16a, 34, nullptr, (float*)d_out, nullptr, HDIM, 1, S15, S15, 3);
}
